// RouteGNN_16681652978020
// MI455X (gfx1250) — matrix operand order repaired and checked by reading, not compiled, not run on hardware
//
#include <hip/hip_runtime.h>
#include <hip/hip_bf16.h>

typedef __attribute__((ext_vector_type(16))) _Float16 v16h;
typedef __attribute__((ext_vector_type(8)))  _Float16 v8h;
typedef __attribute__((ext_vector_type(8)))  float    v8f;
typedef __attribute__((ext_vector_type(4)))  float    v4f;

#define N_NODES 4096
#define F_IN 8
#define H 64
#define KSPLIT 8
#define KCHUNK (N_NODES / KSPLIT)   // 512

// ---------------- encoder: h = relu(X @ W_enc^T + b_enc) ----------------
__global__ __launch_bounds__(256) void k_encoder(const float* __restrict__ X,
                                                 const float* __restrict__ W,
                                                 const float* __restrict__ b,
                                                 float* __restrict__ h) {
  int idx = blockIdx.x * 256 + threadIdx.x;   // N*H threads
  int i = idx >> 6, o = idx & 63;
  const float* xr = X + i * F_IN;
  const float* wr = W + o * F_IN;
  float acc = b[o];
#pragma unroll
  for (int k = 0; k < F_IN; ++k) acc = fmaf(xr[k], wr[k], acc);
  h[idx] = fmaxf(acc, 0.0f);
}

// ------------- transpose + convert: hT[o][i] = (f16) h[i][o] -------------
__global__ __launch_bounds__(256) void k_to_f16_T(const float* __restrict__ h,
                                                  _Float16* __restrict__ hT) {
  int idx = blockIdx.x * 256 + threadIdx.x;   // N*H threads
  int o = idx >> 12;                          // 0..63
  int i = idx & (N_NODES - 1);                // 0..4095
  hT[(size_t)o * N_NODES + i] = (_Float16)h[(size_t)i * H + o];
}

// ---- fragment loaders (inlined) ----
__device__ __forceinline__ void load_a_frag(const float* __restrict__ arow,
                                            int k0, int half, v16h& af) {
  const int ka = k0 + half * 16;
  v4f a0 = *(const v4f*)(arow + ka);
  v4f a1 = *(const v4f*)(arow + ka + 4);
  v4f a2 = *(const v4f*)(arow + ka + 8);
  v4f a3 = *(const v4f*)(arow + ka + 12);
#pragma unroll
  for (int e = 0; e < 4; ++e) {
    af[e]      = (_Float16)a0[e];
    af[e + 4]  = (_Float16)a1[e];
    af[e + 8]  = (_Float16)a2[e];
    af[e + 12] = (_Float16)a3[e];
  }
}

__device__ __forceinline__ void load_b_frags(const _Float16* __restrict__ bcol,
                                             int k0, int half,
                                             v16h& b0, v16h& b1, v16h& b2, v16h& b3) {
  const int kb = k0 + half * 16;
  const _Float16* bp = bcol + kb;
  v8h x0 = *(const v8h*)(bp + 0 * 16 * N_NODES);
  v8h y0 = *(const v8h*)(bp + 0 * 16 * N_NODES + 8);
  v8h x1 = *(const v8h*)(bp + 1 * 16 * N_NODES);
  v8h y1 = *(const v8h*)(bp + 1 * 16 * N_NODES + 8);
  v8h x2 = *(const v8h*)(bp + 2 * 16 * N_NODES);
  v8h y2 = *(const v8h*)(bp + 2 * 16 * N_NODES + 8);
  v8h x3 = *(const v8h*)(bp + 3 * 16 * N_NODES);
  v8h y3 = *(const v8h*)(bp + 3 * 16 * N_NODES + 8);
  b0 = __builtin_shufflevector(x0, y0, 0,1,2,3,4,5,6,7,8,9,10,11,12,13,14,15);
  b1 = __builtin_shufflevector(x1, y1, 0,1,2,3,4,5,6,7,8,9,10,11,12,13,14,15);
  b2 = __builtin_shufflevector(x2, y2, 0,1,2,3,4,5,6,7,8,9,10,11,12,13,14,15);
  b3 = __builtin_shufflevector(x3, y3, 0,1,2,3,4,5,6,7,8,9,10,11,12,13,14,15);
}

// ------- hn_part[chunk] = A[:, chunk] @ h[chunk, :] via f16 WMMA -------
// Block = one 16-row M-tile; its 8 waves each own one K-chunk of 512.
// Software-pipelined: two ping-pong fragment sets so each WMMA's operands
// have been in flight for a full half-iteration (no wait-0 stalls).
__global__ __launch_bounds__(256) void k_gemm_neigh(const float* __restrict__ A,
                                                    const _Float16* __restrict__ hT,
                                                    float* __restrict__ hn_part) {
  const int lane  = threadIdx.x & 31;
  const int chunk = threadIdx.x >> 5;         // k-chunk 0..7
  const int m0    = blockIdx.x * 16;
  const int half  = lane >> 4;
  const int lrow  = lane & 15;

  const float*    arow = A  + (size_t)(m0 + lrow) * N_NODES;
  const _Float16* bcol = hT + (size_t)lrow * N_NODES;

  v8f c0 = {}, c1 = {}, c2 = {}, c3 = {};

  const int kbeg = chunk * KCHUNK;
  const int kend = kbeg + KCHUNK;

  v16h aA, bA0, bA1, bA2, bA3;    // set A (current)
  v16h aB, bB0, bB1, bB2, bB3;    // set B (next)

  load_a_frag(arow, kbeg, half, aA);
  load_b_frags(bcol, kbeg, half, bA0, bA1, bA2, bA3);

  for (int k0 = kbeg; k0 < kend; k0 += 64) {
    const int k1 = k0 + 32;
    // prefetch set B while set A computes
    load_a_frag(arow, k1, half, aB);
    load_b_frags(bcol, k1, half, bB0, bB1, bB2, bB3);
    __builtin_prefetch(arow + k0 + 128, 0, 1);

    c0 = __builtin_amdgcn_wmma_f32_16x16x32_f16(false, aA, false, bA0, (short)0, c0, false, false);
    c1 = __builtin_amdgcn_wmma_f32_16x16x32_f16(false, aA, false, bA1, (short)0, c1, false, false);
    c2 = __builtin_amdgcn_wmma_f32_16x16x32_f16(false, aA, false, bA2, (short)0, c2, false, false);
    c3 = __builtin_amdgcn_wmma_f32_16x16x32_f16(false, aA, false, bA3, (short)0, c3, false, false);

    // refill set A for the next outer iteration (redirect tail to kbeg: in-bounds, unused)
    const int k2 = (k1 + 32 < kend) ? (k1 + 32) : kbeg;
    load_a_frag(arow, k2, half, aA);
    load_b_frags(bcol, k2, half, bA0, bA1, bA2, bA3);

    c0 = __builtin_amdgcn_wmma_f32_16x16x32_f16(false, aB, false, bB0, (short)0, c0, false, false);
    c1 = __builtin_amdgcn_wmma_f32_16x16x32_f16(false, aB, false, bB1, (short)0, c1, false, false);
    c2 = __builtin_amdgcn_wmma_f32_16x16x32_f16(false, aB, false, bB2, (short)0, c2, false, false);
    c3 = __builtin_amdgcn_wmma_f32_16x16x32_f16(false, aB, false, bB3, (short)0, c3, false, false);
  }

  // C/D layout: VGPR r, lane -> C[m0 + r + 8*half][tile*16 + lrow]
  float* part = hn_part + (size_t)chunk * N_NODES * H;
#pragma unroll
  for (int r = 0; r < 8; ++r) {
    float* rowp = part + (size_t)(m0 + r + 8 * half) * H + lrow;
    rowp[0]  = c0[r];
    rowp[16] = c1[r];
    rowp[32] = c2[r];
    rowp[48] = c3[r];
  }
}

// -------- hn = sum over 8 K-chunk partials (vectorized b128) --------
__global__ __launch_bounds__(256) void k_reduce8(const float* __restrict__ part,
                                                 float* __restrict__ hn) {
  int idx4 = (blockIdx.x * 256 + threadIdx.x) * 4;   // N*H/4 threads
  v4f s = *(const v4f*)(part + idx4);
#pragma unroll
  for (int c = 1; c < KSPLIT; ++c) {
    v4f p = *(const v4f*)(part + (size_t)c * N_NODES * H + idx4);
#pragma unroll
    for (int e = 0; e < 4; ++e) s[e] += p[e];
  }
  *(v4f*)(hn + idx4) = s;
}

// -------- h_out = relu([h, hn] @ W^T + b), W is 64x128 row-major --------
__global__ __launch_bounds__(256) void k_combine(const float* __restrict__ h,
                                                 const float* __restrict__ hn,
                                                 const float* __restrict__ W,
                                                 const float* __restrict__ b,
                                                 float* __restrict__ hout) {
  int idx = blockIdx.x * 256 + threadIdx.x;   // N*H threads
  int i = idx >> 6, o = idx & 63;
  const float* wr  = W  + o * (2 * H);
  const float* hr  = h  + (size_t)i * H;
  const float* hnr = hn + (size_t)i * H;
  float acc = b[o];
#pragma unroll 8
  for (int k = 0; k < H; ++k) acc = fmaf(hr[k], wr[k], acc);
#pragma unroll 8
  for (int k = 0; k < H; ++k) acc = fmaf(hnr[k], wr[H + k], acc);
  hout[idx] = fmaxf(acc, 0.0f);
}

// -------- s_i = h @ W_edge[0,:64], s_j = h @ W_edge[0,64:] --------
__global__ __launch_bounds__(256) void k_edge_proj(const float* __restrict__ h,
                                                   const float* __restrict__ We,
                                                   float* __restrict__ s_i,
                                                   float* __restrict__ s_j) {
  int i = blockIdx.x * 256 + threadIdx.x;     // N threads
  const float* hr = h + (size_t)i * H;
  float a = 0.0f, c = 0.0f;
#pragma unroll 8
  for (int k = 0; k < H; ++k) {
    float hv = hr[k];
    a = fmaf(hv, We[k], a);
    c = fmaf(hv, We[H + k], c);
  }
  s_i[i] = a;
  s_j[i] = c;
}

// -------- scores[i][j] = s_i[i] + s_j[j] + b_edge, zero diagonal --------
__global__ __launch_bounds__(256) void k_scores(const float* __restrict__ s_i,
                                                const float* __restrict__ s_j,
                                                const float* __restrict__ b_edge,
                                                float* __restrict__ out) {
  unsigned idx = blockIdx.x * 256 + threadIdx.x;  // N * (N/4) threads
  int i  = idx >> 10;
  int j0 = (idx & 1023) << 2;
  float si = s_i[i] + b_edge[0];
  v4f sj = *(const v4f*)(s_j + j0);
  v4f v;
#pragma unroll
  for (int c = 0; c < 4; ++c) v[c] = (j0 + c == i) ? 0.0f : (si + sj[c]);
  *(v4f*)(out + (size_t)i * N_NODES + j0) = v;
}

extern "C" void kernel_launch(void* const* d_in, const int* in_sizes, int n_in,
                              void* d_out, int out_size, void* d_ws, size_t ws_size,
                              hipStream_t stream) {
  (void)in_sizes; (void)n_in; (void)out_size; (void)ws_size;
  const float* X      = (const float*)d_in[0];
  const float* A      = (const float*)d_in[1];
  const float* W_enc  = (const float*)d_in[2];
  const float* b_enc  = (const float*)d_in[3];
  const float* Wl[3]  = {(const float*)d_in[4], (const float*)d_in[6], (const float*)d_in[8]};
  const float* bl[3]  = {(const float*)d_in[5], (const float*)d_in[7], (const float*)d_in[9]};
  const float* W_edge = (const float*)d_in[10];
  const float* b_edge = (const float*)d_in[11];
  float* out = (float*)d_out;

  // workspace layout (~11.6 MB total)
  char* ws = (char*)d_ws;
  float*    hA      = (float*)ws;                              // 1 MB
  float*    hB      = (float*)(ws + (size_t)(1u << 20));       // 1 MB
  float*    hn_part = (float*)(ws + (size_t)(2u << 20));       // 8 MB (8 x 1 MB)
  float*    hn      = (float*)(ws + (size_t)(10u << 20));      // 1 MB
  _Float16* hT      = (_Float16*)(ws + (size_t)(11u << 20));   // 512 KB
  float*    s_i     = (float*)(ws + (size_t)(11u << 20) + (512u << 10));
  float*    s_j     = s_i + N_NODES;

  const int NH_BLOCKS = (N_NODES * H) / 256;                   // 1024

  k_encoder<<<NH_BLOCKS, 256, 0, stream>>>(X, W_enc, b_enc, hA);

  float* cur = hA;
  float* nxt = hB;
  for (int l = 0; l < 3; ++l) {
    k_to_f16_T<<<NH_BLOCKS, 256, 0, stream>>>(cur, hT);
    k_gemm_neigh<<<N_NODES / 16, 256, 0, stream>>>(A, hT, hn_part);   // 256 blocks x 8 waves
    k_reduce8<<<(N_NODES * H / 4) / 256, 256, 0, stream>>>(hn_part, hn);
    k_combine<<<NH_BLOCKS, 256, 0, stream>>>(cur, hn, Wl[l], bl[l], nxt);
    float* t = cur; cur = nxt; nxt = t;
  }

  k_edge_proj<<<N_NODES / 256, 256, 0, stream>>>(cur, W_edge, s_i, s_j);
  k_scores<<<(N_NODES * (N_NODES / 4)) / 256, 256, 0, stream>>>(s_i, s_j, b_edge, out);
}
